// TFXLNetMainLayer_55336358641796
// MI455X (gfx1250) — matrix operand order repaired and checked by reading, not compiled, not run on hardware
//
#include <hip/hip_runtime.h>
#include <math.h>
#include <stdint.h>

// XLNet 2-layer block for MI455X (gfx1250, wave32).
// GEMMs: async HBM->LDS staging (double buffered) + v_wmma_f32_16x16x32_bf16,
// 2x2 register tiles per wave. Softmax/LN/GELU in fp32 VALU.

#define S_  1024
#define B_  2
#define D_  1024
#define NH  16
#define DH_ 64
#define DI_ 4096
#define R_  2048
#define ND  (NH*DH_)   // 1024
#define TOK (S_*B_)    // 2048
#define BN  (B_*NH)    // 32

#define TM   128       // workgroup tile M
#define TN   64        // workgroup tile N
#define PK   32        // K step (one WMMA depth)
#define PIT  40        // LDS row pitch (elements): 80B rows, 16B aligned, bank-spread

typedef __bf16 bf16;
typedef __attribute__((ext_vector_type(16))) __bf16 v16bf;
typedef __attribute__((ext_vector_type(8)))  __bf16 v8bf;
typedef __attribute__((ext_vector_type(8)))  float  v8f;

// ---- CDNA5 async copy: per-lane 16B HBM -> LDS, tracked by ASYNCcnt --------
__device__ __forceinline__ void async_b128(unsigned lds_byte_off, const void* gaddr) {
    asm volatile("global_load_async_to_lds_b128 %0, %1, off"
                 :: "v"(lds_byte_off), "v"(gaddr) : "memory");
}
__device__ __forceinline__ void wait_async0() {
    asm volatile("s_wait_asynccnt 0" ::: "memory");
}
__device__ __forceinline__ unsigned lds_off(const void* p) {
    // generic pointer into LDS aperture: low 32 bits are the LDS byte offset
    return (unsigned)(uintptr_t)p;
}

// ---------------------------------------------------------------------------
// C[M,N] (f32) = A[M,K] (bf16, row-major, lda) x Bt[N,K]^T (bf16, row-major, ldb)
// Batched via blockIdx.z (element strides zA/zB/zC).
// 256 threads = 8 waves (4 M x 2 N); each wave owns a 32x32 tile (2x2 WMMA).
// Fragment layouts per CDNA5 ISA 7.12.2 (wave32):
//   A 16x32 : lane l holds row (l&15); K chunks at {8h} and {16+8h}, h=l>>4
//   B 32x16 : lane l holds col (l&15); K chunks at {8h} and {16+8h}, same as A
//   C 16x16 : VGPR v -> row v + 8h, lane -> column
// ---------------------------------------------------------------------------
__global__ __launch_bounds__(256)
void k_gemm_bf16_tn(const bf16* __restrict__ A, const bf16* __restrict__ Bt,
                    float* __restrict__ C, int M, int N, int K,
                    int lda, int ldb, int ldc,
                    long long zA, long long zB, long long zC)
{
    __shared__ __align__(16) bf16 lds_a[2][TM * PIT];  // 2*128*40*2 = 20480 B
    __shared__ __align__(16) bf16 lds_b[2][TN * PIT];  // 2* 64*40*2 = 10240 B

    const int z = blockIdx.z;
    A  += (long long)z * zA;
    Bt += (long long)z * zB;
    C  += (long long)z * zC;

    const int tid  = threadIdx.x;
    const int wave = tid >> 5;
    const int lane = tid & 31;
    const int l16  = lane & 15;
    const int half = lane >> 4;
    const int wm   = (wave & 3) * 32;   // wave row block inside tile
    const int wn   = (wave >> 2) * 32;  // wave col block inside tile

    const long long tm0 = (long long)blockIdx.y * TM;
    const long long tn0 = (long long)blockIdx.x * TN;

    // staging assignment: 16B chunks; A = 512 chunks (2/thread), B = 256 (1/thread)
    const int srow = tid >> 2;            // 0..63
    const int scol = (tid & 3) * 8;       // element offset of chunk
    const bf16* gA0 = A  + (tm0 + srow)      * (long long)lda + scol;
    const bf16* gA1 = A  + (tm0 + 64 + srow) * (long long)lda + scol;
    const bf16* gB0 = Bt + (tn0 + srow)      * (long long)ldb + scol;
    unsigned oA0[2], oA1[2], oB0[2];
    #pragma unroll
    for (int p = 0; p < 2; ++p) {
        oA0[p] = lds_off(&lds_a[p][srow * PIT + scol]);
        oA1[p] = lds_off(&lds_a[p][(64 + srow) * PIT + scol]);
        oB0[p] = lds_off(&lds_b[p][srow * PIT + scol]);
    }

    v8f acc00 = {}, acc01 = {}, acc10 = {}, acc11 = {};

    auto stage = [&](int p, int k0) {
        async_b128(oA0[p], gA0 + k0);
        async_b128(oA1[p], gA1 + k0);
        async_b128(oB0[p], gB0 + k0);
        if (k0 + PK < K) {  // warm GL2 one more step ahead
            __builtin_prefetch((const void*)(gA0 + k0 + PK), 0, 2);
            __builtin_prefetch((const void*)(gB0 + k0 + PK), 0, 2);
        }
    };

    stage(0, 0);
    wait_async0();
    __syncthreads();

    int buf = 0;
    for (int k0 = 0; k0 < K; k0 += PK) {
        if (k0 + PK < K) stage(buf ^ 1, k0 + PK);

        const bf16* pA = &lds_a[buf][0];
        const bf16* pB = &lds_b[buf][0];
        // A fragments (rows wm+l16 and wm+16+l16)
        v8bf a0lo = *(const v8bf*)(pA + (wm + l16) * PIT + half * 8);
        v8bf a0hi = *(const v8bf*)(pA + (wm + l16) * PIT + 16 + half * 8);
        v8bf a1lo = *(const v8bf*)(pA + (wm + 16 + l16) * PIT + half * 8);
        v8bf a1hi = *(const v8bf*)(pA + (wm + 16 + l16) * PIT + 16 + half * 8);
        // B fragments (cols wn+l16 and wn+16+l16)
        v8bf b0lo = *(const v8bf*)(pB + (wn + l16) * PIT + half * 8);
        v8bf b0hi = *(const v8bf*)(pB + (wn + l16) * PIT + 16 + half * 8);
        v8bf b1lo = *(const v8bf*)(pB + (wn + 16 + l16) * PIT + half * 8);
        v8bf b1hi = *(const v8bf*)(pB + (wn + 16 + l16) * PIT + 16 + half * 8);
        v16bf a0, a1, b0, b1;
        #pragma unroll
        for (int e = 0; e < 8; ++e) {
            a0[e] = a0lo[e]; a0[e + 8] = a0hi[e];
            a1[e] = a1lo[e]; a1[e + 8] = a1hi[e];
            b0[e] = b0lo[e]; b0[e + 8] = b0hi[e];
            b1[e] = b1lo[e]; b1[e + 8] = b1hi[e];
        }
        acc00 = __builtin_amdgcn_wmma_f32_16x16x32_bf16(false, a0, false, b0, (short)0, acc00, false, false);
        acc01 = __builtin_amdgcn_wmma_f32_16x16x32_bf16(false, a0, false, b1, (short)0, acc01, false, false);
        acc10 = __builtin_amdgcn_wmma_f32_16x16x32_bf16(false, a1, false, b0, (short)0, acc10, false, false);
        acc11 = __builtin_amdgcn_wmma_f32_16x16x32_bf16(false, a1, false, b1, (short)0, acc11, false, false);

        wait_async0();      // our async writes into buf^1 have landed
        __syncthreads();    // everyone done reading buf / writing buf^1
        buf ^= 1;
    }

    float* c00 = C + (tm0 + wm + half * 8) * (long long)ldc + tn0 + wn + l16;
    float* c10 = c00 + 16LL * ldc;
    #pragma unroll
    for (int v = 0; v < 8; ++v) {
        c00[(long long)v * ldc]      = acc00[v];
        c00[(long long)v * ldc + 16] = acc01[v];
        c10[(long long)v * ldc]      = acc10[v];
        c10[(long long)v * ldc + 16] = acc11[v];
    }
}

// ------------------------- elementwise helpers -----------------------------
__global__ void k_cvt_bf16(const float* __restrict__ s, bf16* __restrict__ d,
                           long long n)
{
    long long i = (long long)blockIdx.x * 256 + threadIdx.x;
    if (i < n) d[i] = (bf16)s[i];
}

// src[Rr,Cc] (f32) -> dst[Cc,Rr] (bf16)
__global__ void k_transpose_cvt(const float* __restrict__ src, bf16* __restrict__ dst,
                                int Rr, int Cc, long long total)
{
    long long i = (long long)blockIdx.x * 256 + threadIdx.x;
    if (i >= total) return;
    int r = (int)(i / Cc), c = (int)(i % Cc);
    dst[(long long)c * Rr + r] = (bf16)src[i];
}

__global__ void k_add_bias_bf16(const float* __restrict__ x, const float* __restrict__ bias,
                                bf16* __restrict__ y, int Nc, long long total)
{
    long long i = (long long)blockIdx.x * 256 + threadIdx.x;
    if (i >= total) return;
    y[i] = (bf16)(x[i] + bias[(int)(i % Nc)]);
}

// vh f32 [S,B,N,DH] -> vhT bf16 [B,N,DH,S]   (for prob x V with contiguous B-frags)
__global__ void k_vhT(const float* __restrict__ vh, bf16* __restrict__ vt, long long total)
{
    long long i = (long long)blockIdx.x * 256 + threadIdx.x;
    if (i >= total) return;
    int d = (int)(i & 63);
    long long t = i >> 6;
    int n = (int)(t & (NH - 1)); t >>= 4;
    int b = (int)(t & (B_ - 1));
    long long j = t >> 1;
    vt[((((long long)b * NH + n) * DH_ + d) << 10) + j] = (bf16)vh[i];
}

// segment scores: ef[z=b*NH+n][i][s] = sum_d (qh[i,b,n,d]+rs_bias[n,d]) * segE[s,n,d]
__global__ void k_ef(const float* __restrict__ qh, const float* __restrict__ rsb,
                     const float* __restrict__ segE, float* __restrict__ ef)
{
    int gid = blockIdx.x * blockDim.x + threadIdx.x;
    if (gid >= BN * S_) return;
    int z = gid / S_, i = gid % S_;
    int b = z / NH, n = z % NH;
    const float* q   = qh + (((long long)i * B_ + b) * NH + n) * DH_;
    const float* bs  = rsb + n * DH_;
    const float* s0p = segE + (0 * NH + n) * DH_;
    const float* s1p = segE + (1 * NH + n) * DH_;
    float s0 = 0.f, s1 = 0.f;
    #pragma unroll 4
    for (int d = 0; d < DH_; ++d) {
        float t = q[d] + bs[d];
        s0 += t * s0p[d];
        s1 += t * s1p[d];
    }
    ef[(long long)gid * 2 + 0] = s0;
    ef[(long long)gid * 2 + 1] = s1;
}

// score = (ac + bd_shifted + ef_sel)*scale - 1e30*mask; row softmax -> prob bf16.
// rel_shift: bd_shifted[i,j] = bd_full[i, S + j - i]  (q=S, r=2S reshape identity)
__global__ __launch_bounds__(256)
void k_score_softmax(const float* __restrict__ ac, const float* __restrict__ bdf,
                     const float* __restrict__ ef, const float* __restrict__ amask,
                     const float* __restrict__ segm, bf16* __restrict__ prob)
{
    const int i = blockIdx.x;
    const int z = blockIdx.y;
    const int b = z / NH;
    const int tid = threadIdx.x;
    const float scale = 0.125f;  // 1/sqrt(64)
    const float* acr = ac  + ((long long)z * S_ + i) * S_;
    const float* bdr = bdf + ((long long)z * S_ + i) * R_ + (S_ - i);
    const float e0 = ef[((long long)z * S_ + i) * 2 + 0];
    const float e1 = ef[((long long)z * S_ + i) * 2 + 1];
    __shared__ float red[256];
    float sc[4];
    float mx = -3.0e38f;
    #pragma unroll
    for (int t = 0; t < 4; ++t) {
        const int j = tid + t * 256;
        const float* sm = segm + (((long long)i * S_ + j) * B_ + b) * 2;
        float efv = sm[0] * e0 + sm[1] * e1;
        float s = (acr[j] + bdr[j] + efv) * scale - 1.0e30f * amask[i * S_ + j];
        sc[t] = s;
        mx = fmaxf(mx, s);
    }
    red[tid] = mx; __syncthreads();
    for (int o = 128; o > 0; o >>= 1) {
        if (tid < o) red[tid] = fmaxf(red[tid], red[tid + o]);
        __syncthreads();
    }
    mx = red[0]; __syncthreads();
    float sum = 0.f;
    #pragma unroll
    for (int t = 0; t < 4; ++t) { sc[t] = __expf(sc[t] - mx); sum += sc[t]; }
    red[tid] = sum; __syncthreads();
    for (int o = 128; o > 0; o >>= 1) {
        if (tid < o) red[tid] += red[tid + o];
        __syncthreads();
    }
    const float inv = 1.0f / red[0];
    bf16* pr = prob + ((long long)z * S_ + i) * S_;
    #pragma unroll
    for (int t = 0; t < 4; ++t) pr[tid + t * 256] = (bf16)(sc[t] * inv);
}

// LayerNorm of (xa + xb + optional bias) over Dm, one block per row.
__global__ __launch_bounds__(256)
void k_residual_ln(const float* __restrict__ xa, const float* __restrict__ xb,
                   const float* __restrict__ bias, const float* __restrict__ g,
                   const float* __restrict__ bb, float* __restrict__ o, int Dm)
{
    const int row = blockIdx.x;
    const int tid = threadIdx.x;
    const float* pa = xa + (long long)row * Dm;
    const float* pb = xb + (long long)row * Dm;
    __shared__ float red[256];
    float s = 0.f;
    for (int c = tid; c < Dm; c += 256)
        s += pa[c] + pb[c] + (bias ? bias[c] : 0.f);
    red[tid] = s; __syncthreads();
    for (int off = 128; off > 0; off >>= 1) {
        if (tid < off) red[tid] += red[tid + off];
        __syncthreads();
    }
    const float mean = red[0] / Dm; __syncthreads();
    float v = 0.f;
    for (int c = tid; c < Dm; c += 256) {
        float t = pa[c] + pb[c] + (bias ? bias[c] : 0.f) - mean;
        v += t * t;
    }
    red[tid] = v; __syncthreads();
    for (int off = 128; off > 0; off >>= 1) {
        if (tid < off) red[tid] += red[tid + off];
        __syncthreads();
    }
    const float inv = rsqrtf(red[0] / Dm + 1e-12f);
    for (int c = tid; c < Dm; c += 256) {
        float t = (pa[c] + pb[c] + (bias ? bias[c] : 0.f) - mean) * inv;
        o[(long long)row * Dm + c] = t * g[c] + bb[c];
    }
}

// exact GELU on (x + bias), convert to bf16
__global__ void k_bias_gelu_bf16(const float* __restrict__ x, const float* __restrict__ bias,
                                 bf16* __restrict__ y, int Nc, long long total)
{
    long long i = (long long)blockIdx.x * 256 + threadIdx.x;
    if (i >= total) return;
    float t = x[i] + bias[(int)(i % Nc)];
    y[i] = (bf16)(0.5f * t * (1.0f + erff(t * 0.70710678118654752f)));
}

// ---------------------------------------------------------------------------
extern "C" void kernel_launch(void* const* d_in, const int* in_sizes, int n_in,
                              void* d_out, int out_size, void* d_ws, size_t ws_size,
                              hipStream_t stream)
{
    (void)in_sizes; (void)n_in; (void)out_size; (void)ws_size;
    const float* h    = (const float*)d_in[0];   // [S,B,D]
    const float* pos  = (const float*)d_in[1];   // [R,B,D]
    const float* mask = (const float*)d_in[2];   // [S,S,1,1]
    const float* segm = (const float*)d_in[3];   // [S,S,B,2]
    const float* Wq   = (const float*)d_in[4];   // [L,D,N,DH]
    const float* Wk   = (const float*)d_in[5];
    const float* Wv   = (const float*)d_in[6];
    const float* Wo   = (const float*)d_in[7];
    const float* Wr   = (const float*)d_in[8];
    const float* rwb  = (const float*)d_in[9];   // [L,N,DH]
    const float* rrb  = (const float*)d_in[10];
    const float* rsb  = (const float*)d_in[11];
    const float* segE = (const float*)d_in[12];  // [L,2,N,DH]
    const float* ln1g = (const float*)d_in[13];
    const float* ln1b = (const float*)d_in[14];
    const float* fw1  = (const float*)d_in[15];  // [L,D,DI]
    const float* fb1  = (const float*)d_in[16];
    const float* fw2  = (const float*)d_in[17];  // [L,DI,D]
    const float* fb2  = (const float*)d_in[18];
    const float* ln2g = (const float*)d_in[19];
    const float* ln2b = (const float*)d_in[20];
    float* out = (float*)d_out;

    char* base = (char*)d_ws;
    size_t off = 0;
    auto alloc = [&](size_t bytes) -> void* {
        off = (off + 255) & ~(size_t)255;
        void* p = base + off;
        off += bytes;
        return p;
    };

    bf16* hbf   = (bf16*)alloc((size_t)TOK * D_ * 2);
    bf16* posbf = (bf16*)alloc((size_t)R_ * B_ * D_ * 2);
    bf16 *wqt[2], *wkt[2], *wvt[2], *wrt[2], *wob[2], *w1t[2], *w2t[2];
    for (int l = 0; l < 2; ++l) {
        wqt[l] = (bf16*)alloc((size_t)D_ * ND * 2);
        wkt[l] = (bf16*)alloc((size_t)D_ * ND * 2);
        wvt[l] = (bf16*)alloc((size_t)D_ * ND * 2);
        wrt[l] = (bf16*)alloc((size_t)D_ * ND * 2);
        wob[l] = (bf16*)alloc((size_t)D_ * ND * 2);
        w1t[l] = (bf16*)alloc((size_t)D_ * DI_ * 2);
        w2t[l] = (bf16*)alloc((size_t)DI_ * D_ * 2);
    }
    float* qh   = (float*)alloc((size_t)TOK * ND * 4);
    bf16*  qw   = (bf16*)alloc((size_t)TOK * ND * 2);
    bf16*  qr   = (bf16*)alloc((size_t)TOK * ND * 2);
    bf16*  khbf = (bf16*)alloc((size_t)TOK * ND * 2);
    bf16*  vhT  = (bf16*)alloc((size_t)BN * DH_ * S_ * 2);
    bf16*  krbf = (bf16*)alloc((size_t)R_ * B_ * ND * 2);
    float* efb  = (float*)alloc((size_t)BN * S_ * 2 * 4);
    float* ac   = (float*)alloc((size_t)BN * S_ * S_ * 4);
    float* bdf  = (float*)alloc((size_t)BN * S_ * R_ * 4);
    bf16*  prob = (bf16*)alloc((size_t)BN * S_ * S_ * 2);
    float* vec  = (float*)alloc((size_t)TOK * ND * 4);
    bf16*  vecb = (bf16*)alloc((size_t)TOK * ND * 2);
    float* xv   = (float*)alloc((size_t)TOK * D_ * 4);
    bf16*  xbf  = (bf16*)alloc((size_t)TOK * D_ * 2);
    float* tmpb = (float*)alloc((size_t)TOK * DI_ * 4);   // reused f32 GEMM out
    bf16*  fbf  = (bf16*)alloc((size_t)TOK * DI_ * 2);
    float* f2   = (float*)alloc((size_t)TOK * D_ * 4);
    float* h1   = (float*)alloc((size_t)TOK * D_ * 4);

    auto cvt = [&](const float* s, bf16* d, long long n) {
        k_cvt_bf16<<<(int)((n + 255) / 256), 256, 0, stream>>>(s, d, n);
    };
    auto trans = [&](const float* s, bf16* d, int Rr, int Cc) {
        long long n = (long long)Rr * Cc;
        k_transpose_cvt<<<(int)((n + 255) / 256), 256, 0, stream>>>(s, d, Rr, Cc, n);
    };
    auto gemm = [&](const bf16* Ap, const bf16* Bp, float* Cp, int M, int N, int K,
                    int lda, int ldb, int ldc,
                    long long zA, long long zB, long long zC, int batch) {
        dim3 grid(N / TN, M / TM, batch);
        k_gemm_bf16_tn<<<grid, 256, 0, stream>>>(Ap, Bp, Cp, M, N, K,
                                                 lda, ldb, ldc, zA, zB, zC);
    };

    // ---- one-time conversions ----
    cvt(h, hbf, (long long)TOK * D_);
    cvt(pos, posbf, (long long)R_ * B_ * D_);
    for (int l = 0; l < 2; ++l) {
        trans(Wq + (long long)l * D_ * ND, wqt[l], D_, ND);
        trans(Wk + (long long)l * D_ * ND, wkt[l], D_, ND);
        trans(Wv + (long long)l * D_ * ND, wvt[l], D_, ND);
        trans(Wr + (long long)l * D_ * ND, wrt[l], D_, ND);
        cvt(Wo + (long long)l * D_ * ND, wob[l], (long long)D_ * ND);  // [h, nd] natural
        trans(fw1 + (long long)l * D_ * DI_, w1t[l], D_, DI_);
        trans(fw2 + (long long)l * DI_ * D_, w2t[l], DI_, D_);
    }

    const float* hcur = h;
    for (int l = 0; l < 2; ++l) {
        long long nTok = (long long)TOK * ND;
        // Q projection, then biased copies + segment scores
        gemm(hbf, wqt[l], qh, TOK, ND, D_, D_, D_, ND, 0, 0, 0, 1);
        k_add_bias_bf16<<<(int)((nTok + 255) / 256), 256, 0, stream>>>(
            qh, rwb + l * ND, qw, ND, nTok);
        k_add_bias_bf16<<<(int)((nTok + 255) / 256), 256, 0, stream>>>(
            qh, rrb + l * ND, qr, ND, nTok);
        k_ef<<<(BN * S_ + 255) / 256, 256, 0, stream>>>(
            qh, rsb + l * ND, segE + (long long)l * 2 * ND, efb);
        // K, V, R projections
        gemm(hbf, wkt[l], tmpb, TOK, ND, D_, D_, D_, ND, 0, 0, 0, 1);
        cvt(tmpb, khbf, nTok);
        gemm(hbf, wvt[l], tmpb, TOK, ND, D_, D_, D_, ND, 0, 0, 0, 1);
        k_vhT<<<(int)((nTok + 255) / 256), 256, 0, stream>>>(tmpb, vhT, nTok);
        gemm(posbf, wrt[l], tmpb, R_ * B_, ND, D_, D_, D_, ND, 0, 0, 0, 1);
        cvt(tmpb, krbf, (long long)R_ * B_ * ND);
        // per-head scores (batched over z = b*NH + n via strides of DH)
        gemm(qw, khbf, ac, S_, S_, DH_, B_ * ND, B_ * ND, S_,
             DH_, DH_, (long long)S_ * S_, BN);
        gemm(qr, krbf, bdf, S_, R_, DH_, B_ * ND, B_ * ND, R_,
             DH_, DH_, (long long)S_ * R_, BN);
        k_score_softmax<<<dim3(S_, BN), 256, 0, stream>>>(ac, bdf, efb, mask, segm, prob);
        // prob x V  -> vec [i,b,n,d]
        gemm(prob, vhT, vec, S_, DH_, S_, S_, S_, B_ * ND,
             (long long)S_ * S_, (long long)DH_ * S_, DH_, BN);
        cvt(vec, vecb, nTok);
        // output projection + residual LN1
        gemm(vecb, wob[l], tmpb, TOK, D_, ND, ND, ND, D_, 0, 0, 0, 1);
        k_residual_ln<<<TOK, 256, 0, stream>>>(tmpb, hcur, (const float*)nullptr,
                                               ln1g + l * D_, ln1b + l * D_, xv, D_);
        cvt(xv, xbf, (long long)TOK * D_);
        // FFN
        gemm(xbf, w1t[l], tmpb, TOK, DI_, D_, D_, D_, DI_, 0, 0, 0, 1);
        long long nF = (long long)TOK * DI_;
        k_bias_gelu_bf16<<<(int)((nF + 255) / 256), 256, 0, stream>>>(
            tmpb, fb1 + l * DI_, fbf, DI_, nF);
        gemm(fbf, w2t[l], f2, TOK, D_, DI_, DI_, DI_, D_, 0, 0, 0, 1);
        float* hn = (l == 0) ? h1 : out;
        k_residual_ln<<<TOK, 256, 0, stream>>>(f2, xv, fb2 + l * D_,
                                               ln2g + l * D_, ln2b + l * D_, hn, D_);
        if (l == 0) {
            cvt(hn, hbf, (long long)TOK * D_);
            hcur = hn;
        }
    }
}
